// CrossAttnFusion_87582973100534
// MI455X (gfx1250) — matrix operand order repaired and checked by reading, not compiled, not run on hardware
//
#include <hip/hip_runtime.h>
#include <hip/hip_bf16.h>

typedef __attribute__((ext_vector_type(16))) _Float16 v16h;
typedef __attribute__((ext_vector_type(8)))  _Float16 v8h;
typedef __attribute__((ext_vector_type(8)))  float    v8f;

#define NTOK 49
#define NP   64
#define CDIM 384
#define HD   32
#define NHEAD 12
#define KT   12              // K tiles of 32 over CDIM
#define NTILES 24            // N tiles of 16 over CDIM
#define FRAG (KT*NTILES*32*16)   // halves per repacked weight matrix

__device__ __forceinline__ v8f zv8() {
    v8f z;
#pragma unroll
    for (int i = 0; i < 8; ++i) z[i] = 0.f;
    return z;
}

__device__ __forceinline__ v8f wmma16(v16h a, v16h b, v8f c) {
    return __builtin_amdgcn_wmma_f32_16x16x32_f16(false, a, false, b, (short)0, c, false, false);
}

// A-fragment (16x32 f16, lane=row, K packed {klo..klo+15} contiguous)
// assembled from a row-major LDS half array; p points at row_base + kstep*32 + klo.
__device__ __forceinline__ v16h load_a_lds(const _Float16* p) {
    v8h lo = *(const v8h*)p;
    v8h hi = *(const v8h*)(p + 8);
    v16h a;
#pragma unroll
    for (int i = 0; i < 8; ++i) { a[i] = lo[i]; a[8 + i] = hi[i]; }
    return a;
}

// Same, but returns zeros for pad rows (row >= NTOK).
__device__ __forceinline__ v16h load_a_guard(const _Float16* p, int ok) {
    v16h a;
#pragma unroll
    for (int i = 0; i < 16; ++i) a[i] = (_Float16)0.f;
    if (ok) {
        v8h lo = *(const v8h*)p;
        v8h hi = *(const v8h*)(p + 8);
#pragma unroll
        for (int i = 0; i < 8; ++i) { a[i] = lo[i]; a[8 + i] = hi[i]; }
    }
    return a;
}

// ---------------- Kernel 1: mod = cond @ mod_w + mod_b  (32 x 768) ----------------
__global__ void mod_kernel(const float* __restrict__ cond, const float* __restrict__ mod_w,
                           const float* __restrict__ mod_b, float* __restrict__ mod) {
    int idx = blockIdx.x * 256 + threadIdx.x;
    if (idx >= 32 * 768) return;
    int b = idx / 768, j = idx % 768;
    float s = mod_b[j];
    for (int k = 0; k < 128; ++k) s += cond[b * 128 + k] * mod_w[k * 768 + j];
    mod[idx] = s;
}

// ---------------- Kernel 2: repack weights into WMMA B-fragment order -------------
// layout: frag[((kt*NTILES + nt)*32 + lane)*16 + i]; lane holds column nt*16+(lane&15),
// K = kt*32 + (lane<16?0:16) + i  (16 contiguous K values per lane).
__global__ void repack_kernel(const float* __restrict__ qw, const float* __restrict__ kvw,
                              const float* __restrict__ pw,
                              _Float16* __restrict__ qf, _Float16* __restrict__ kf,
                              _Float16* __restrict__ vf, _Float16* __restrict__ pf) {
    int idx = blockIdx.x * 256 + threadIdx.x;
    if (idx >= 4 * FRAG) return;
    int mat = idx / FRAG;
    int rem = idx % FRAG;
    int i = rem & 15;
    int lane = (rem >> 4) & 31;
    int tile = rem >> 9;
    int nt = tile % NTILES;
    int kt = tile / NTILES;
    int n = lane & 15;
    int K = kt * 32 + ((lane < 16) ? 0 : 16) + i;
    int Nc = nt * 16 + n;
    float v;
    _Float16* dst;
    if (mat == 0)      { v = qw[K * 384 + Nc];        dst = qf; }
    else if (mat == 1) { v = kvw[K * 768 + Nc];       dst = kf; }
    else if (mat == 2) { v = kvw[K * 768 + 384 + Nc]; dst = vf; }
    else               { v = pw[K * 384 + Nc];        dst = pf; }
    dst[rem] = (_Float16)v;
}

// ---------------- Kernel 3: fused per-window attention ----------------------------
__global__ __launch_bounds__(256) void attn_kernel(
    const float* __restrict__ x_tok, const float* __restrict__ z_tok,
    const float* __restrict__ gamma_x, const float* __restrict__ beta_x,
    const float* __restrict__ gamma_z, const float* __restrict__ beta_z,
    const float* __restrict__ mod, const float* __restrict__ rel_bias,
    const float* __restrict__ q_b, const float* __restrict__ kv_b,
    const float* __restrict__ proj_b,
    const _Float16* __restrict__ qw_f, const _Float16* __restrict__ kw_f,
    const _Float16* __restrict__ vw_f, const _Float16* __restrict__ pw_f,
    float* __restrict__ out) {

    // ~114 KB LDS: CDNA5 WGP has 320 KB, so stage whole normalized window in f16.
    __shared__ alignas(32) _Float16 x_s[NTOK * CDIM];   // normalized+AdaLN x window
    __shared__ alignas(32) _Float16 z_s[NTOK * CDIM];   // normalized z window
    __shared__ alignas(32) _Float16 q_s[NP * HD];
    __shared__ alignas(32) _Float16 k_s[NP * HD];
    __shared__ alignas(32) _Float16 vT_s[HD * NP];      // v transposed: [dim][token]
    __shared__ alignas(32) _Float16 oh_s[NP * HD];
    __shared__ alignas(32) _Float16 at_s[NP * NP];
    __shared__ float sc_s[NP * NP];                     // scores; overlaid affine params in phase 0
    __shared__ unsigned off_s[NP];

    float* Gx  = sc_s;          // [384]  gamma_x*(1+d_gamma)
    float* Bxs = sc_s + 384;    // [384]  beta_x + d_beta
    float* Gz  = sc_s + 768;    // [384]
    float* Bzs = sc_s + 1152;   // [384]

    const int tid  = threadIdx.x;
    const int wave = tid >> 5;
    const int lane = tid & 31;
    const int widx = blockIdx.x;
    const int b  = widx >> 6;
    const int wy = (widx >> 3) & 7;
    const int wx = widx & 7;

    // token offsets for this window
    if (tid < NP) {
        int r = (tid < NTOK) ? tid : (NTOK - 1);
        int y  = wy * 7 + r / 7;
        int xx = wx * 7 + r % 7;
        off_s[tid] = ((unsigned)b * 3136u + (unsigned)(y * 56 + xx)) * 384u;
    }
    // per-window affine params (AdaLN for x, plain LN for z)
    for (int t = tid; t < CDIM; t += 256) {
        float dg = mod[b * 768 + t];
        float db = mod[b * 768 + 384 + t];
        Gx[t]  = gamma_x[t] * (1.f + dg);
        Bxs[t] = beta_x[t] + db;
        Gz[t]  = gamma_z[t];
        Bzs[t] = beta_z[t];
    }
    __syncthreads();

    // LayerNorm stats + normalize + stage to LDS as f16 (one wave per row)
    for (int r = wave; r < NTOK; r += 8) {
        unsigned base = off_s[r];
        float vx[12], vz[12];
        float s = 0.f, sq = 0.f;
#pragma unroll
        for (int i = 0; i < 12; ++i) {
            float v = x_tok[base + lane + 32 * i];
            vx[i] = v; s += v; sq += v * v;
        }
        for (int o = 16; o > 0; o >>= 1) { s += __shfl_xor(s, o, 32); sq += __shfl_xor(sq, o, 32); }
        {
            float mu = s / 384.f, var = sq / 384.f - mu * mu;
            float rs = rsqrtf(var + 1e-6f);
#pragma unroll
            for (int i = 0; i < 12; ++i) {
                int k = lane + 32 * i;
                x_s[r * CDIM + k] = (_Float16)((vx[i] - mu) * rs * Gx[k] + Bxs[k]);
            }
        }
        s = 0.f; sq = 0.f;
#pragma unroll
        for (int i = 0; i < 12; ++i) {
            float v = z_tok[base + lane + 32 * i];
            vz[i] = v; s += v; sq += v * v;
        }
        for (int o = 16; o > 0; o >>= 1) { s += __shfl_xor(s, o, 32); sq += __shfl_xor(sq, o, 32); }
        {
            float mu = s / 384.f, var = sq / 384.f - mu * mu;
            float rs = rsqrtf(var + 1e-5f);
#pragma unroll
            for (int i = 0; i < 12; ++i) {
                int k = lane + 32 * i;
                z_s[r * CDIM + k] = (_Float16)((vz[i] - mu) * rs * Gz[k] + Bzs[k]);
            }
        }
    }
    __syncthreads();

    const float scale = 0.17677669529663687f;   // 32^-0.5
    const int klo = (lane < 16) ? 0 : 16;       // A-fragment K base within 32
    const int bklo = (lane < 16) ? 0 : 16;      // B-fragment K base within 32
    const int nlo = lane & 15;
    const int mhalf = (lane >> 4) << 3;         // D rows 0..7 or 8..15

    // persistent proj accumulators: 12 output tiles per wave (4x24 grid / 8 waves)
    v8f pacc[12];
#pragma unroll
    for (int j = 0; j < 12; ++j) pacc[j] = zv8();

    for (int h = 0; h < NHEAD; ++h) {
        // ---- Q/K/V projections for this head (12 tasks: {q,k,v} x 4 m-tiles) ----
        for (int task = wave; task < 12; task += 8) {
            int mat = task >> 2;       // 0=q 1=k 2=v
            int mt  = task & 3;
            int mrow = mt * 16 + nlo;
            int rok = mrow < NTOK;
            const _Float16* rowp = ((mat == 0) ? x_s : z_s) + mrow * CDIM;
            const _Float16* wf = (mat == 0) ? qw_f : ((mat == 1) ? kw_f : vw_f);
            int nt0 = h * 2;
            v8f acc0 = zv8(), acc1 = zv8();
            for (int kt = 0; kt < KT; ++kt) {
                v16h a = load_a_guard(rowp + kt * 32 + klo, rok);
                const _Float16* fb = wf + (((kt * NTILES + nt0) * 32) + lane) * 16;
                v16h b0 = *(const v16h*)fb;
                v16h b1 = *(const v16h*)(fb + 512);
                acc0 = wmma16(a, b0, acc0);
                acc1 = wmma16(a, b1, acc1);
            }
            int mb = mt * 16 + mhalf;
            if (mat == 0) {
                float bb0 = q_b[h * 32 + nlo], bb1 = q_b[h * 32 + 16 + nlo];
#pragma unroll
                for (int r = 0; r < 8; ++r) {
                    q_s[(mb + r) * 32 + nlo]      = (_Float16)((acc0[r] + bb0) * scale);
                    q_s[(mb + r) * 32 + 16 + nlo] = (_Float16)((acc1[r] + bb1) * scale);
                }
            } else if (mat == 1) {
                float bb0 = kv_b[h * 32 + nlo], bb1 = kv_b[h * 32 + 16 + nlo];
#pragma unroll
                for (int r = 0; r < 8; ++r) {
                    k_s[(mb + r) * 32 + nlo]      = (_Float16)(acc0[r] + bb0);
                    k_s[(mb + r) * 32 + 16 + nlo] = (_Float16)(acc1[r] + bb1);
                }
            } else {
                float bb0 = kv_b[384 + h * 32 + nlo], bb1 = kv_b[384 + h * 32 + 16 + nlo];
                v8h t0, t1;
#pragma unroll
                for (int r = 0; r < 8; ++r) {
                    t0[r] = (_Float16)(acc0[r] + bb0);
                    t1[r] = (_Float16)(acc1[r] + bb1);
                }
                *(v8h*)&vT_s[nlo * 64 + mb]        = t0;   // v stored transposed: [dim][token]
                *(v8h*)&vT_s[(16 + nlo) * 64 + mb] = t1;
            }
        }
        __syncthreads();

        // ---- scores = (q*scale) @ k^T : 16 tiles, K=32 -> one WMMA each ----
        for (int t4 = wave; t4 < 16; t4 += 8) {
            int mt = t4 >> 2, nt = t4 & 3;
            v16h a = load_a_lds(&q_s[(mt * 16 + nlo) * 32 + klo]);
            v16h bf = *(const v16h*)&k_s[(nt * 16 + nlo) * 32 + bklo];  // k row-major == B-frag
            v8f acc = wmma16(a, bf, zv8());
            int n  = nt * 16 + nlo;
            int mb = mt * 16 + mhalf;
#pragma unroll
            for (int r = 0; r < 8; ++r) sc_s[(mb + r) * 64 + n] = acc[r];
        }
        __syncthreads();

        // ---- softmax rows with relative-position bias ----
        if (tid < NP) {
            int r = tid;
            if (r < NTOK) {
                int yi = r / 7, xi = r % 7;
                float mx = -1e30f;
                for (int c = 0; c < NTOK; ++c) {
                    int idx = (yi - c / 7 + 6) * 13 + (xi - c % 7 + 6);
                    float s = sc_s[r * 64 + c] + rel_bias[idx * NHEAD + h];
                    sc_s[r * 64 + c] = s;
                    mx = fmaxf(mx, s);
                }
                float sum = 0.f;
                for (int c = 0; c < NTOK; ++c) {
                    float e = __expf(sc_s[r * 64 + c] - mx);
                    sc_s[r * 64 + c] = e;
                    sum += e;
                }
                float rinv = 1.f / sum;
                for (int c = 0; c < NP; ++c)
                    at_s[r * 64 + c] = (c < NTOK) ? (_Float16)(sc_s[r * 64 + c] * rinv)
                                                  : (_Float16)0.f;
            } else {
                for (int c = 0; c < NP; ++c) at_s[r * 64 + c] = (_Float16)0.f;
            }
        }
        __syncthreads();

        // ---- out_h = attn @ v : 8 tiles (one per wave), K=64 -> 2 WMMA ----
        {
            int mt = wave >> 1, nt = wave & 1;
            v8f acc = zv8();
#pragma unroll
            for (int ks = 0; ks < 2; ++ks) {
                v16h a  = load_a_lds(&at_s[(mt * 16 + nlo) * 64 + ks * 32 + klo]);
                v16h bf = *(const v16h*)&vT_s[(nt * 16 + nlo) * 64 + ks * 32 + bklo];
                acc = wmma16(a, bf, acc);
            }
            int n  = nt * 16 + nlo;
            int mb = mt * 16 + mhalf;
#pragma unroll
            for (int r = 0; r < 8; ++r) oh_s[(mb + r) * 32 + n] = (_Float16)acc[r];
        }
        __syncthreads();

        // ---- proj accumulation: out_h (K-slice h) @ proj_w, in registers ----
#pragma unroll
        for (int j = 0; j < 12; ++j) {
            int tidx = wave + j * 8;         // 0..95 over 4x24 tile grid
            int mt = tidx / NTILES, nt = tidx % NTILES;
            v16h a  = load_a_lds(&oh_s[(mt * 16 + nlo) * 32 + klo]);
            v16h bf = *(const v16h*)(pw_f + (((h * NTILES + nt) * 32) + lane) * 16);
            pacc[j] = wmma16(a, bf, pacc[j]);
        }
        __syncthreads();
    }

    // ---- epilogue: + proj_b + residual, scatter back to token layout ----
#pragma unroll
    for (int j = 0; j < 12; ++j) {
        int tidx = wave + j * 8;
        int mt = tidx / NTILES, nt = tidx % NTILES;
        int n  = nt * 16 + nlo;
        int mb = mt * 16 + mhalf;
        float pb = proj_b[n];
#pragma unroll
        for (int r = 0; r < 8; ++r) {
            int m = mb + r;
            if (m < NTOK) {
                unsigned base = off_s[m];
                out[base + n] = pacc[j][r] + pb + x_tok[base + n];
            }
        }
    }
}

extern "C" void kernel_launch(void* const* d_in, const int* in_sizes, int n_in,
                              void* d_out, int out_size, void* d_ws, size_t ws_size,
                              hipStream_t stream) {
    const float* x_tok   = (const float*)d_in[0];
    const float* z_tok   = (const float*)d_in[1];
    const float* cond    = (const float*)d_in[2];
    const float* gamma_x = (const float*)d_in[3];
    const float* beta_x  = (const float*)d_in[4];
    const float* mod_w   = (const float*)d_in[5];
    const float* mod_b   = (const float*)d_in[6];
    const float* gamma_z = (const float*)d_in[7];
    const float* beta_z  = (const float*)d_in[8];
    const float* rel_tab = (const float*)d_in[9];
    const float* q_w     = (const float*)d_in[10];
    const float* q_b     = (const float*)d_in[11];
    const float* kv_w    = (const float*)d_in[12];
    const float* kv_b    = (const float*)d_in[13];
    const float* proj_w  = (const float*)d_in[14];
    const float* proj_b  = (const float*)d_in[15];
    float* out = (float*)d_out;

    // workspace layout: mod (32*768 f32 = 98304 B), then 4 repacked f16 weight mats
    float* ws_mod = (float*)d_ws;
    _Float16* qf = (_Float16*)((char*)d_ws + 98304);
    _Float16* kf = qf + FRAG;
    _Float16* vf = kf + FRAG;
    _Float16* pf = vf + FRAG;

    mod_kernel<<<96, 256, 0, stream>>>(cond, mod_w, mod_b, ws_mod);
    repack_kernel<<<(4 * FRAG + 255) / 256, 256, 0, stream>>>(q_w, kv_w, proj_w, qf, kf, vf, pf);
    attn_kernel<<<2048, 256, 0, stream>>>(x_tok, z_tok, gamma_x, beta_x, gamma_z, beta_z,
                                          ws_mod, rel_tab, q_b, kv_b, proj_b,
                                          qf, kf, vf, pf, out);
}
